// OfficialMambaClassifier_14362370638406
// MI455X (gfx1250) — matrix operand order repaired and checked by reading, not compiled, not run on hardware
//
#include <hip/hip_runtime.h>
#include <hip/hip_bf16.h>
#include <math.h>

// ---------------------------------------------------------------------------
// Mamba classifier forward, MI455X (gfx1250, wave32).
//   B=32, L=1024, D_MODEL=128, D_INNER=256, D_STATE=16, D_CONV=4, DT_RANK=8
// in_proj and out_proj/fc/pool are algebraically collapsed (rank-1 structure);
// the real GEMM (x_proj: 32768x48x256) runs on v_wmma_f32_16x16x32_f16.
// ---------------------------------------------------------------------------

#define BB 32
#define LL 1024
#define DM 128
#define DI 256
#define NS 16
#define DC 4
#define DTR 8
#define MM (BB * LL)        // 32768 tokens
#define NPAD 48             // 40 -> 48 (3 tiles of 16)

typedef _Float16 v16h __attribute__((ext_vector_type(16)));
typedef _Float16 v8h  __attribute__((ext_vector_type(8)));
typedef float    v8f  __attribute__((ext_vector_type(8)));

__device__ __forceinline__ float silu_f(float x) {
    return x * (1.0f / (1.0f + __expf(-x)));
}
__device__ __forceinline__ float softplus_f(float x) {
    return (x > 20.0f) ? x : log1pf(__expf(x));
}

// ---------------------------------------------------------------------------
// K0: tiny precompute (1 block).
//   u1[e] = dot(W_in[e,:], W_emb[:,0])   e < 512
//   u0[e] = dot(W_in[e,:], b_emb)
//   v[e]  = sum_dm W_fc[0,dm] * W_out[dm,e]   e < 256
//   Wx16  = f16(W_x) padded to 48x256 (rows 40..47 = 0)
// ---------------------------------------------------------------------------
__global__ void k0_precompute(const float* __restrict__ W_emb,
                              const float* __restrict__ b_emb,
                              const float* __restrict__ W_in,
                              const float* __restrict__ W_x,
                              const float* __restrict__ W_out,
                              const float* __restrict__ W_fc,
                              float* __restrict__ u1,
                              float* __restrict__ u0,
                              float* __restrict__ vv,
                              _Float16* __restrict__ wx16) {
    const int t = threadIdx.x;                 // 512 threads
    if (t < 2 * DI) {
        float s1 = 0.f, s0 = 0.f;
        const float* row = W_in + t * DM;
        for (int d = 0; d < DM; ++d) {
            s1 += row[d] * W_emb[d];
            s0 += row[d] * b_emb[d];
        }
        u1[t] = s1;
        u0[t] = s0;
    }
    if (t < DI) {
        float s = 0.f;
        for (int dm = 0; dm < DM; ++dm) s += W_fc[dm] * W_out[dm * DI + t];
        vv[t] = s;
    }
    // W_x -> f16, padded to NPAD rows
    for (int idx = t; idx < NPAD * DI; idx += blockDim.x) {
        int j = idx / DI, c = idx % DI;
        float w = (j < (DTR + 2 * NS)) ? W_x[j * DI + c] : 0.0f;
        wx16[idx] = (_Float16)w;
    }
}

// ---------------------------------------------------------------------------
// K1: u[b,l,d] = silu(conv_b[d] + sum_k conv_w[k,d] * xssm(b, l-3+k, d))
//     xssm(b,l',d) = x[b,l'] * u1[d] + u0[d]  (zero for l' < 0: causal pad)
// Stored as f16, row-major [M x 256] -> WMMA A operand of the x_proj GEMM.
// ---------------------------------------------------------------------------
__global__ void k1_compute_u(const float* __restrict__ x,
                             const float* __restrict__ conv_w,
                             const float* __restrict__ conv_b,
                             const float* __restrict__ u1,
                             const float* __restrict__ u0,
                             _Float16* __restrict__ u16) {
    const int bl = blockIdx.x;                 // 0..MM-1
    const int d  = threadIdx.x;                // 0..255
    const int b  = bl >> 10;
    const int l  = bl & (LL - 1);
    const float a1 = u1[d], a0 = u0[d];
    float acc = conv_b[d];
#pragma unroll
    for (int k = 0; k < DC; ++k) {
        int lk = l - (DC - 1) + k;
        float xs = (lk >= 0) ? (x[b * LL + lk] * a1 + a0) : 0.0f;
        acc += conv_w[k * DI + d] * xs;
    }
    u16[bl * DI + d] = (_Float16)silu_f(acc);
}

// ---------------------------------------------------------------------------
// K2: x_db = u @ W_x^T via WMMA.  M=32768, N=48, K=256.
// One wave per 16x16 output tile; 8 x v_wmma_f32_16x16x32_f16 along K.
// Fragment layouts used here (wave32; A/B in contiguous k order, not the ISA table order):
//   A 16x32 f16 : lanes 0-15 -> M=lane, halves[0..15]=K 0..15
//                 lanes16-31 -> M=lane-16, halves[0..15]=K 16..31
//   B 32x16 f16 : lane n=lane&15, halves[i] = B[kb + i][n], kb = lane<16 ? 0 : 16
//   C/D f32     : vgpr r -> M = r + (lane<16 ? 0 : 8), N = lane&15
// Results scattered: j<8 -> dt_raw, j<24 -> Bs, j<40 -> Cs, j>=40 dropped.
// ---------------------------------------------------------------------------
__global__ void k2_xproj_wmma(const _Float16* __restrict__ u16,
                              const _Float16* __restrict__ wx16,
                              float* __restrict__ dtr,
                              float* __restrict__ Bs,
                              float* __restrict__ Cs) {
    const int lane = threadIdx.x & 31;
    const int wid  = (blockIdx.x * blockDim.x + threadIdx.x) >> 5;  // global wave
    const int mt   = wid / 3;          // 0..2047
    const int nt   = wid % 3;          // 0..2
    const int m0   = mt * 16;
    const int n0   = nt * 16;

    const int arow  = lane & 15;
    const int koffA = (lane < 16) ? 0 : 16;  // half-chunk offset within K-tile
    const int koffB = (lane < 16) ? 0 : 16;
    const int ncol  = lane & 15;

    v8f c = {};
#pragma unroll
    for (int kt = 0; kt < DI / 32; ++kt) {
        const int kb = kt * 32;
        // A fragment: two contiguous 8-half chunks
        const _Float16* ap = u16 + (size_t)(m0 + arow) * DI + kb + koffA;
        v8h a_lo = *(const v8h*)(ap);
        v8h a_hi = *(const v8h*)(ap + 8);
        v16h a = __builtin_shufflevector(a_lo, a_hi,
                                         0, 1, 2, 3, 4, 5, 6, 7,
                                         8, 9, 10, 11, 12, 13, 14, 15);
        // B fragment: 16 contiguous halves of one W_x row (K-major)
        const _Float16* bp = wx16 + (size_t)(n0 + ncol) * DI + kb + koffB;
        v16h bfrag = *(const v16h*)(bp);

        if (kt + 1 < DI / 32)
            __builtin_prefetch(ap + 32, 0, 3);   // global_prefetch next A chunk

        c = __builtin_amdgcn_wmma_f32_16x16x32_f16(
                /*neg_a=*/false, a, /*neg_b=*/false, bfrag,
                /*c_mod=*/(short)0, c, /*reuse_a=*/false, /*reuse_b=*/false);
    }

    // scatter C: vgpr r -> row m0 + r + (lane<16?0:8), col j = n0 + (lane&15)
    const int j = n0 + ncol;
    const int mbase = m0 + ((lane < 16) ? 0 : 8);
#pragma unroll
    for (int r = 0; r < 8; ++r) {
        const int m = mbase + r;
        const float val = c[r];
        if (j < DTR)                 dtr[(size_t)m * DTR + j];
        if (j < DTR)                 dtr[(size_t)m * DTR + j] = val;
        else if (j < DTR + NS)       Bs[(size_t)m * NS + (j - DTR)] = val;
        else if (j < DTR + 2 * NS)   Cs[(size_t)m * NS + (j - DTR - NS)] = val;
        // j in [40,48): padding, dropped
    }
}

// ---------------------------------------------------------------------------
// K3: fused selective scan + gate + out_proj/fc/pool collapse + sigmoid.
// One block per batch b; thread d owns channel d's 16-state recurrence.
// u and z are recomputed on the fly from scalar x[b,l] (rank-1 in_proj).
// acc_d += y[b,l,d] * v[d]; block-reduce -> logits[b] -> sigmoid.
// ---------------------------------------------------------------------------
__global__ void k3_scan(const float* __restrict__ x,
                        const float* __restrict__ conv_w,
                        const float* __restrict__ conv_b,
                        const float* __restrict__ W_dt,
                        const float* __restrict__ b_dt,
                        const float* __restrict__ A_log,
                        const float* __restrict__ Dvec,
                        const float* __restrict__ b_fc,
                        const float* __restrict__ u1,
                        const float* __restrict__ u0,
                        const float* __restrict__ vv,
                        const float* __restrict__ dtr,
                        const float* __restrict__ BsA,
                        const float* __restrict__ CsA,
                        float* __restrict__ out) {
    const int b = blockIdx.x;       // 0..31
    const int d = threadIdx.x;      // 0..255
    __shared__ float red[DI];

    // per-channel constants
    const float a1 = u1[d],      a0 = u0[d];
    const float z1 = u1[DI + d], z0 = u0[DI + d];
    float cw[DC];
#pragma unroll
    for (int k = 0; k < DC; ++k) cw[k] = conv_w[k * DI + d];
    const float cb = conv_b[d];
    float wdt[DTR];
#pragma unroll
    for (int j = 0; j < DTR; ++j) wdt[j] = W_dt[d * DTR + j];
    const float bdt = b_dt[d];
    float Arow[NS];
#pragma unroll
    for (int n = 0; n < NS; ++n) Arow[n] = -__expf(A_log[d * NS + n]);
    const float Dd = Dvec[d];
    const float vd = vv[d];

    float h[NS];
#pragma unroll
    for (int n = 0; n < NS; ++n) h[n] = 0.0f;

    // sliding window of x scalars (taps l-3..l)
    float xw0 = 0.f, xw1 = 0.f, xw2 = 0.f, xw3 = 0.f;
    float acc = 0.0f;
    const float* xb = x + b * LL;

    for (int l = 0; l < LL; ++l) {
        const float xc = xb[l];
        xw0 = xw1; xw1 = xw2; xw2 = xw3; xw3 = xc;

        // u = silu(causal depthwise conv of xssm)
        float conv = cb;
        conv += (l >= 3) ? cw[0] * (xw0 * a1 + a0) : 0.0f;
        conv += (l >= 2) ? cw[1] * (xw1 * a1 + a0) : 0.0f;
        conv += (l >= 1) ? cw[2] * (xw2 * a1 + a0) : 0.0f;
        conv += cw[3] * (xw3 * a1 + a0);
        const float uu = silu_f(conv);

        // dt = softplus(dt_raw . W_dt[d] + b_dt[d])   (broadcast 8 floats)
        const float* dtp = dtr + (size_t)(b * LL + l) * DTR;
        float dv = bdt;
#pragma unroll
        for (int j = 0; j < DTR; ++j) dv += dtp[j] * wdt[j];
        dv = softplus_f(dv);

        const float dtu = dv * uu;
        const float* Bp = BsA + (size_t)(b * LL + l) * NS;
        const float* Cp = CsA + (size_t)(b * LL + l) * NS;
        float y = 0.0f;
#pragma unroll
        for (int n = 0; n < NS; ++n) {
            const float dA = __expf(dv * Arow[n]);
            h[n] = dA * h[n] + dtu * Bp[n];
            y += h[n] * Cp[n];
        }
        y += uu * Dd;
        y *= silu_f(xc * z1 + z0);      // gate with silu(z), z recomputed
        acc += y * vd;                   // folded out_proj . fc
    }

    red[d] = acc;
    __syncthreads();
#pragma unroll
    for (int s = DI / 2; s > 0; s >>= 1) {
        if (d < s) red[d] += red[d + s];
        __syncthreads();
    }
    if (d == 0) {
        const float logit = red[0] * (1.0f / (float)LL) + b_fc[0];
        out[b] = 1.0f / (1.0f + __expf(-logit));
    }
}

// ---------------------------------------------------------------------------
extern "C" void kernel_launch(void* const* d_in, const int* in_sizes, int n_in,
                              void* d_out, int out_size, void* d_ws, size_t ws_size,
                              hipStream_t stream) {
    const float* x      = (const float*)d_in[0];
    const float* W_emb  = (const float*)d_in[1];
    const float* b_emb  = (const float*)d_in[2];
    const float* W_in   = (const float*)d_in[3];
    const float* conv_w = (const float*)d_in[4];
    const float* conv_b = (const float*)d_in[5];
    const float* W_x    = (const float*)d_in[6];
    const float* W_dt   = (const float*)d_in[7];
    const float* b_dt   = (const float*)d_in[8];
    const float* A_log  = (const float*)d_in[9];
    const float* Dvec   = (const float*)d_in[10];
    /* W_out */           // d_in[11] folded into v
    const float* W_out  = (const float*)d_in[11];
    const float* W_fc   = (const float*)d_in[12];
    const float* b_fc   = (const float*)d_in[13];
    float* out = (float*)d_out;

    // workspace layout (all 16B-aligned)
    char* ws = (char*)d_ws;
    float*    u1   = (float*)(ws + 0);            // 512
    float*    u0   = (float*)(ws + 2048);         // 512
    float*    vv   = (float*)(ws + 4096);         // 256
    _Float16* wx16 = (_Float16*)(ws + 8192);      // 48*256 halves
    float*    dtr  = (float*)(ws + 32768);                      // MM*8
    float*    Bs   = dtr + (size_t)MM * DTR;                    // MM*16
    float*    Cs   = Bs  + (size_t)MM * NS;                     // MM*16
    _Float16* u16  = (_Float16*)(Cs + (size_t)MM * NS);         // MM*256 halves

    k0_precompute<<<1, 512, 0, stream>>>(W_emb, b_emb, W_in, W_x, W_out, W_fc,
                                         u1, u0, vv, wx16);
    k1_compute_u<<<MM, DI, 0, stream>>>(x, conv_w, conv_b, u1, u0, u16);
    // 2048 M-tiles * 3 N-tiles = 6144 waves; 8 waves (256 thr) per block
    k2_xproj_wmma<<<6144 / 8, 256, 0, stream>>>(u16, wx16, dtr, Bs, Cs);
    k3_scan<<<BB, DI, 0, stream>>>(x, conv_w, conv_b, W_dt, b_dt, A_log, Dvec,
                                   b_fc, u1, u0, vv, dtr, Bs, Cs, out);
}
